// S4Layer_69887707841219
// MI455X (gfx1250) — matrix operand order repaired and checked by reading, not compiled, not run on hardware
//
#include <hip/hip_runtime.h>

typedef __bf16 bf16;
typedef __attribute__((ext_vector_type(16))) __bf16 v16bf;
typedef __attribute__((ext_vector_type(8)))  float  v8f;

#define HIDDEN 1024
#define STATE  64
#define BATCH  16
#define SEQ    2048
#define NROWS  (BATCH * SEQ)   // 32768
#define CHUNK  64
#define NCHUNK (SEQ / CHUNK)   // 32

// ---------------------------------------------------------------------------
// prep (1 block): dt = exp(log_dt), dt_mean; A_bar = expm(A*dt_mean) via
// Taylor (8 terms, ||X||~0.02); M = A_bar^T; ML = M^64 (6 squarings);
// stage bf16 weights: W1T[n][h] = B[n][h]*dt[h], Cbf[h][n] = C[h][n].
// ---------------------------------------------------------------------------
__global__ __launch_bounds__(256) void s4_prep(
    const float* __restrict__ A, const float* __restrict__ Bw,
    const float* __restrict__ Cw, const float* __restrict__ log_dt,
    float* __restrict__ M, float* __restrict__ ML,
    bf16* __restrict__ W1T, bf16* __restrict__ Cbf)
{
  __shared__ float sm[16384];          // 64 KB
  const int tid = threadIdx.x;

  // dt and its mean
  float lsum = 0.f;
  for (int i = 0; i < 4; ++i) {
    int h = tid + 256 * i;
    float d = __expf(log_dt[h]);
    sm[h] = d;                         // dt lives in sm[0..1023]
    lsum += d;
  }
  __syncthreads();
  sm[1024 + tid] = lsum;
  __syncthreads();
  for (int s = 128; s > 0; s >>= 1) {
    if (tid < s) sm[1024 + tid] += sm[1024 + tid + s];
    __syncthreads();
  }
  const float dtm = sm[1024] * (1.0f / 1024.0f);

  // stage bf16 weights (reads dt from LDS)
  for (int idx = tid; idx < HIDDEN * STATE; idx += 256) {
    int h = idx & (HIDDEN - 1);
    W1T[idx] = (bf16)(Bw[idx] * sm[h]);   // (N=64, H=1024), k-contig
    Cbf[idx] = (bf16)(Cw[idx]);           // (H=1024, N=64), k-contig
  }
  __syncthreads();

  // expm via Taylor in LDS: X, P, E, T each 64x64
  float* X = sm;
  float* P = sm + 4096;
  float* E = sm + 8192;
  float* T = sm + 12288;
  for (int i = 0; i < 16; ++i) {
    int idx = tid + 256 * i;
    float xv = A[idx] * dtm;
    X[idx] = xv; P[idx] = xv;
    E[idx] = xv + (((idx % 65) == 0) ? 1.0f : 0.0f);   // I + X
  }
  __syncthreads();
  for (int k = 2; k <= 8; ++k) {
    float invk = 1.0f / (float)k;
    for (int i = 0; i < 16; ++i) {
      int idx = tid + 256 * i;
      int r = idx >> 6, c = idx & 63;
      float a = 0.f;
      for (int l = 0; l < 64; ++l) a += P[(r << 6) + l] * X[(l << 6) + c];
      T[idx] = a * invk;
    }
    __syncthreads();
    for (int i = 0; i < 16; ++i) {
      int idx = tid + 256 * i;
      P[idx] = T[idx];
      E[idx] += T[idx];
    }
    __syncthreads();
  }
  // M = E^T (so scan step is s_new[n] = sum_k s[k]*M[k][n]); Q (in X) = M
  for (int i = 0; i < 16; ++i) {
    int idx = tid + 256 * i;
    int r = idx >> 6, c = idx & 63;
    float v = E[(c << 6) + r];
    M[idx] = v;
    X[idx] = v;
  }
  __syncthreads();
  // ML = M^64 by repeated squaring
  for (int s = 0; s < 6; ++s) {
    for (int i = 0; i < 16; ++i) {
      int idx = tid + 256 * i;
      int r = idx >> 6, c = idx & 63;
      float a = 0.f;
      for (int l = 0; l < 64; ++l) a += X[(r << 6) + l] * X[(l << 6) + c];
      P[idx] = a;
    }
    __syncthreads();
    for (int i = 0; i < 16; ++i) { int idx = tid + 256 * i; X[idx] = P[idx]; }
    __syncthreads();
  }
  for (int i = 0; i < 16; ++i) { int idx = tid + 256 * i; ML[idx] = X[idx]; }
}

// ---------------------------------------------------------------------------
// A-fragment for v_wmma_f32_16x16x32_bf16 built from an f32 row (row = lane&15)
// lane<16: comps 0..15 <- K kb..kb+15
// lane>=16: comps 0..15 <- K kb+16..kb+31
// ---------------------------------------------------------------------------
__device__ inline v16bf load_a_frag_f32(const float* __restrict__ rowp,
                                        int kb, int hi)
{
  const float4* p0 = (const float4*)(rowp + kb + 16 * hi);
  const float4* p1 = (const float4*)(rowp + kb + 8 + 16 * hi);
  float4 f0 = p0[0], f1 = p0[1], f2 = p1[0], f3 = p1[1];
  v16bf a;
  a[0] = (bf16)f0.x; a[1] = (bf16)f0.y; a[2]  = (bf16)f0.z; a[3]  = (bf16)f0.w;
  a[4] = (bf16)f1.x; a[5] = (bf16)f1.y; a[6]  = (bf16)f1.z; a[7]  = (bf16)f1.w;
  a[8] = (bf16)f2.x; a[9] = (bf16)f2.y; a[10] = (bf16)f2.z; a[11] = (bf16)f2.w;
  a[12]= (bf16)f3.x; a[13]= (bf16)f3.y; a[14] = (bf16)f3.z; a[15] = (bf16)f3.w;
  return a;
}

// ---------------------------------------------------------------------------
// GEMM1: u[r][n] = sum_h x[r][h] * W1T[n][h]   (32768 x 1024 x 64)
// 8 waves/block, each wave: 16 rows x 64 n, K loop of 32.
// All 4 B-fragments are loaded into locals before the WMMA burst so the
// backend can clause the 8 b128 loads and wait incrementally per WMMA.
// ---------------------------------------------------------------------------
__global__ __launch_bounds__(256) void s4_gemm1(
    const float* __restrict__ x, const bf16* __restrict__ W1T,
    float* __restrict__ u)
{
  const int tid  = threadIdx.x;
  const int wid  = tid >> 5;
  const int lane = tid & 31;
  const int l15  = lane & 15;
  const int hi   = lane >> 4;
  const int rbase = blockIdx.x * 128 + wid * 16;
  const float* xr = x + (size_t)(rbase + l15) * HIDDEN;
  const bf16* wp = W1T + (size_t)l15 * HIDDEN + 16 * hi;

  v8f acc[4] = {};
  for (int kb = 0; kb < HIDDEN; kb += 32) {
    v16bf b[4];
#pragma unroll
    for (int j = 0; j < 4; ++j)
      b[j] = *(const v16bf*)(wp + (size_t)(j * 16) * HIDDEN + kb);
    v16bf a = load_a_frag_f32(xr, kb, hi);
#pragma unroll
    for (int j = 0; j < 4; ++j)
      acc[j] = __builtin_amdgcn_wmma_f32_16x16x32_bf16(
          false, a, false, b[j], (short)0, acc[j], false, false);
  }
#pragma unroll
  for (int j = 0; j < 4; ++j)
#pragma unroll
    for (int r = 0; r < 8; ++r) {
      int row = rbase + r + 8 * hi;
      u[(size_t)row * STATE + j * 16 + l15] = acc[j][r];
    }
}

// ---------------------------------------------------------------------------
// Chunked linear scan, phase A: per chunk, local scan from zero state;
// emit only the chunk-final 16x64 state. f32 VALU, S + M in LDS.
// ---------------------------------------------------------------------------
__global__ __launch_bounds__(256) void s4_scanA(
    const float* __restrict__ u, const float* __restrict__ M,
    float* __restrict__ Slocend)
{
  __shared__ float Ms[4096];
  __shared__ float S[1024];
  const int tid = threadIdx.x;
  const int c = blockIdx.x;
  for (int i = 0; i < 16; ++i) Ms[tid + 256 * i] = M[tid + 256 * i];
  for (int i = 0; i < 4; ++i)  S[tid + 256 * i] = 0.f;
  __syncthreads();
  const int n = tid & 63;
  int boff[4];
#pragma unroll
  for (int i = 0; i < 4; ++i) boff[i] = ((tid >> 6) + 4 * i) << 6;

  for (int step = 0; step < CHUNK; ++step) {
    const int t = c * CHUNK + step;
    float acc[4];
#pragma unroll
    for (int i = 0; i < 4; ++i) {
      int b = (boff[i] >> 6);
      acc[i] = u[(size_t)(b * SEQ + t) * STATE + n];
    }
    for (int k = 0; k < 64; ++k) {
      float m = Ms[(k << 6) + n];
#pragma unroll
      for (int i = 0; i < 4; ++i) acc[i] += S[boff[i] + k] * m;
    }
    __syncthreads();
#pragma unroll
    for (int i = 0; i < 4; ++i) S[tid + 256 * i] = acc[i];
    __syncthreads();
  }
  for (int i = 0; i < 4; ++i)
    Slocend[(size_t)c * 1024 + tid + 256 * i] = S[tid + 256 * i];
}

// ---------------------------------------------------------------------------
// Phase B (1 block): carries_in[c] = carries_in[c-1]*M^64 + Slocend[c-1]
// ---------------------------------------------------------------------------
__global__ __launch_bounds__(256) void s4_scanB(
    const float* __restrict__ Slocend, const float* __restrict__ ML,
    float* __restrict__ carries)
{
  __shared__ float Ms[4096];
  __shared__ float S[1024];
  const int tid = threadIdx.x;
  for (int i = 0; i < 16; ++i) Ms[tid + 256 * i] = ML[tid + 256 * i];
  for (int i = 0; i < 4; ++i)  S[tid + 256 * i] = 0.f;
  __syncthreads();
  const int n = tid & 63;
  int boff[4];
#pragma unroll
  for (int i = 0; i < 4; ++i) boff[i] = ((tid >> 6) + 4 * i) << 6;

  for (int c = 0; c < NCHUNK; ++c) {
    for (int i = 0; i < 4; ++i)
      carries[(size_t)c * 1024 + tid + 256 * i] = S[tid + 256 * i];
    float acc[4];
#pragma unroll
    for (int i = 0; i < 4; ++i)
      acc[i] = Slocend[(size_t)c * 1024 + tid + 256 * i];
    for (int k = 0; k < 64; ++k) {
      float m = Ms[(k << 6) + n];
#pragma unroll
      for (int i = 0; i < 4; ++i) acc[i] += S[boff[i] + k] * m;
    }
    __syncthreads();
#pragma unroll
    for (int i = 0; i < 4; ++i) S[tid + 256 * i] = acc[i];
    __syncthreads();
  }
}

// ---------------------------------------------------------------------------
// Phase C: re-run each chunk seeded with its carry; write states over u.
// ---------------------------------------------------------------------------
__global__ __launch_bounds__(256) void s4_scanC(
    float* __restrict__ u, const float* __restrict__ M,
    const float* __restrict__ carries)
{
  __shared__ float Ms[4096];
  __shared__ float S[1024];
  const int tid = threadIdx.x;
  const int c = blockIdx.x;
  for (int i = 0; i < 16; ++i) Ms[tid + 256 * i] = M[tid + 256 * i];
  for (int i = 0; i < 4; ++i)
    S[tid + 256 * i] = carries[(size_t)c * 1024 + tid + 256 * i];
  __syncthreads();
  const int n = tid & 63;
  int boff[4];
#pragma unroll
  for (int i = 0; i < 4; ++i) boff[i] = ((tid >> 6) + 4 * i) << 6;

  for (int step = 0; step < CHUNK; ++step) {
    const int t = c * CHUNK + step;
    size_t uoff[4];
    float acc[4];
#pragma unroll
    for (int i = 0; i < 4; ++i) {
      int b = (boff[i] >> 6);
      uoff[i] = (size_t)(b * SEQ + t) * STATE + n;
      acc[i] = u[uoff[i]];
    }
    for (int k = 0; k < 64; ++k) {
      float m = Ms[(k << 6) + n];
#pragma unroll
      for (int i = 0; i < 4; ++i) acc[i] += S[boff[i] + k] * m;
    }
    __syncthreads();
#pragma unroll
    for (int i = 0; i < 4; ++i) S[tid + 256 * i] = acc[i];
    __syncthreads();
#pragma unroll
    for (int i = 0; i < 4; ++i) u[uoff[i]] = acc[i];   // states in place
  }
}

// ---------------------------------------------------------------------------
// GEMM2 + epilogue: y[r][h] = sum_n states[r][n]*C[h][n] + D[h]*x[r][h]
// 8 waves/block: block = 16 rows x 1024 h; wave = 16 x 128; K = 64 (2 steps).
// B-fragments batched into locals before the WMMA burst (clause + staged waits).
// ---------------------------------------------------------------------------
__global__ __launch_bounds__(256) void s4_gemm2(
    const float* __restrict__ st, const bf16* __restrict__ Cbf,
    const float* __restrict__ Dv, const float* __restrict__ x,
    float* __restrict__ out)
{
  const int tid  = threadIdx.x;
  const int wid  = tid >> 5;
  const int lane = tid & 31;
  const int l15  = lane & 15;
  const int hi   = lane >> 4;
  const int rbase  = blockIdx.x * 16;
  const int hbase0 = wid * 128;
  const float* sr = st + (size_t)(rbase + l15) * STATE;
  const bf16* cp = Cbf + (size_t)(hbase0 + l15) * STATE + 16 * hi;

  v8f acc[8] = {};
#pragma unroll
  for (int kk = 0; kk < 2; ++kk) {
    int kb = kk * 32;
    v16bf b[8];
#pragma unroll
    for (int j = 0; j < 8; ++j)
      b[j] = *(const v16bf*)(cp + (size_t)(j * 16) * STATE + kb);
    v16bf a = load_a_frag_f32(sr, kb, hi);
#pragma unroll
    for (int j = 0; j < 8; ++j)
      acc[j] = __builtin_amdgcn_wmma_f32_16x16x32_bf16(
          false, a, false, b[j], (short)0, acc[j], false, false);
  }
#pragma unroll
  for (int j = 0; j < 8; ++j) {
    int h = hbase0 + j * 16 + l15;
    float dh = Dv[h];
#pragma unroll
    for (int r = 0; r < 8; ++r) {
      size_t ro = (size_t)(rbase + r + 8 * hi) * HIDDEN + h;
      out[ro] = acc[j][r] + dh * x[ro];
    }
  }
}

// ---------------------------------------------------------------------------
extern "C" void kernel_launch(void* const* d_in, const int* in_sizes, int n_in,
                              void* d_out, int out_size, void* d_ws, size_t ws_size,
                              hipStream_t stream)
{
  (void)in_sizes; (void)n_in; (void)out_size; (void)ws_size;
  const float* x      = (const float*)d_in[0];
  const float* A      = (const float*)d_in[1];
  const float* Bw     = (const float*)d_in[2];
  const float* Cw     = (const float*)d_in[3];
  const float* Dv     = (const float*)d_in[4];
  const float* log_dt = (const float*)d_in[5];
  float* out = (float*)d_out;

  // workspace carve (~8.6 MiB total)
  char* ws = (char*)d_ws;
  float* u       = (float*)ws;  ws += (size_t)NROWS * STATE * sizeof(float);   // 8 MiB
  float* M       = (float*)ws;  ws += 4096 * sizeof(float);
  float* ML      = (float*)ws;  ws += 4096 * sizeof(float);
  float* Slocend = (float*)ws;  ws += (size_t)NCHUNK * 1024 * sizeof(float);
  float* carries = (float*)ws;  ws += (size_t)NCHUNK * 1024 * sizeof(float);
  bf16*  W1T     = (bf16*)ws;   ws += (size_t)HIDDEN * STATE * sizeof(bf16);
  bf16*  Cbf     = (bf16*)ws;

  s4_prep <<<1,           256, 0, stream>>>(A, Bw, Cw, log_dt, M, ML, W1T, Cbf);
  s4_gemm1<<<NROWS / 128, 256, 0, stream>>>(x, W1T, u);
  s4_scanA<<<NCHUNK,      256, 0, stream>>>(u, M, Slocend);
  s4_scanB<<<1,           256, 0, stream>>>(Slocend, ML, carries);
  s4_scanC<<<NCHUNK,      256, 0, stream>>>(u, M, carries);
  s4_gemm2<<<NROWS / 16,  256, 0, stream>>>(u, Cbf, Dv, x, out);
}
